// CausalSelfAttention_78331613544603
// MI455X (gfx1250) — matrix operand order repaired and checked by reading, not compiled, not run on hardware
//
#include <hip/hip_runtime.h>

// ---------------------------------------------------------------------------
// CDNA5 (gfx1250, wave32) causal self-attention forward.
//   1) fp32 -> bf16 conversion of activations + weights (one pass)
//   2) QKV GEMM      : bf16 x bf16 -> fp32 acc, bf16 out   (v_wmma_f32_16x16x32_bf16)
//                      128x64 tile, double-buffered LDS filled with
//                      global_load_async_to_lds_b128 (ASYNCcnt-tracked)
//   3) flash-attn    : QK^T, online softmax, P*V, all WMMA bf16
//   4) out-proj GEMM : bf16 x bf16 -> fp32 out (d_out)
// Workspace: ~272 MB of bf16 staging in d_ws.
// ---------------------------------------------------------------------------

typedef __attribute__((ext_vector_type(16))) __bf16 v16bf;
typedef __attribute__((ext_vector_type(8)))  float  v8f;

union Frag16 {
  v16bf v;
  unsigned short u[16];
  unsigned int   d[8];
};

constexpr int SEQ   = 2048;
constexpr int BATCH = 2;
constexpr int HID   = 2048;
constexpr int NH    = 16;
constexpr int HD    = 256;              // per-head dim
constexpr int TOK   = SEQ * BATCH;      // 4096 tokens
constexpr int E2    = 2 * HID;          // 4096 input dim
constexpr int F6    = 6 * HID;          // 12288 qkv dim

__device__ __forceinline__ unsigned short f2bf(float f) {
  unsigned int u = __builtin_bit_cast(unsigned int, f);
  u += 0x7FFFu + ((u >> 16) & 1u);      // round-to-nearest-even
  return (unsigned short)(u >> 16);
}

// ---- CDNA5 async global->LDS copies (ASYNCcnt), CDNA5 ISA --
__device__ __forceinline__ void async_ld_b128(unsigned lds_off, const void* g) {
  asm volatile("global_load_async_to_lds_b128 %0, %1, off"
               :: "v"(lds_off), "v"((unsigned long long)(uintptr_t)g)
               : "memory");
}
__device__ __forceinline__ void async_ld_b64(unsigned lds_off, const void* g) {
  asm volatile("global_load_async_to_lds_b64 %0, %1, off"
               :: "v"(lds_off), "v"((unsigned long long)(uintptr_t)g)
               : "memory");
}
__device__ __forceinline__ void wait_async0() {
  asm volatile("s_wait_asynccnt 0x0" ::: "memory");
}

__global__ void cvt_f32_bf16(const float* __restrict__ in,
                             unsigned short* __restrict__ out, int n) {
  for (int i = blockIdx.x * blockDim.x + threadIdx.x; i < n;
       i += gridDim.x * blockDim.x)
    out[i] = f2bf(in[i]);
}

// ---------------------------------------------------------------------------
// GEMM: C[M][N] = A[M][K] * B[N][K]^T   (A,B bf16 row-major K-contiguous)
// Block = 128 threads (4 waves), tile 128x64, K-step 32, double-buffered LDS.
// Wave w computes rows [w*32, w*32+32) x all 64 cols -> 8 WMMA accumulators.
// ---------------------------------------------------------------------------
constexpr int KSTEP = 32;
constexpr int LDT   = 40;               // padded LDS row stride (ushorts), 80B

template <bool OUT_BF16>
__global__ __launch_bounds__(128)
void gemm_bf16(const unsigned short* __restrict__ A,
               const unsigned short* __restrict__ B,
               void* __restrict__ Cv, int M, int N, int K) {
  __shared__ __align__(16) unsigned short Asub[2][128][LDT];
  __shared__ __align__(16) unsigned short Bsub[2][64][LDT];

  const int tid  = threadIdx.x;
  const int lane = tid & 31;
  const int wave = tid >> 5;
  const int hlf  = (lane >> 4) & 1;     // which 16-lane half
  const int l16  = lane & 15;
  const int mbase = blockIdx.y * 128;
  const int nbase = blockIdx.x * 64;

  v8f acc[8] = {};                       // [j*2+m]: rows wave*32+m*16, cols j*16

  // async tile fill: A 128x32 (4x b128/thread), B 64x32 (2x b128/thread)
  auto issue_tile = [&](int k0, int buf) {
#pragma unroll
    for (int j = 0; j < 4; ++j) {
      const unsigned lds = (unsigned)(uintptr_t)&Asub[buf][tid][8 * j];
      async_ld_b128(lds, A + (size_t)(mbase + tid) * K + k0 + 8 * j);
    }
#pragma unroll
    for (int j = 0; j < 2; ++j) {
      const int c   = tid * 2 + j;
      const int row = c >> 2;
      const int cc  = (c & 3) * 8;
      const unsigned lds = (unsigned)(uintptr_t)&Bsub[buf][row][cc];
      async_ld_b128(lds, B + (size_t)(nbase + row) * K + k0 + cc);
    }
  };

  const int nk = K / KSTEP;
  issue_tile(0, 0);
  wait_async0();
  __syncthreads();

  for (int i = 0; i < nk; ++i) {
    const int buf = i & 1;
    if (i + 1 < nk) issue_tile((i + 1) * KSTEP, buf ^ 1);

    // A fragments (16x32 bf16, ISA layout):
    //  lanes 0-15 : row M=l16, pairs p0-3 -> K 0..7,  p4-7 -> K 16..23
    //  lanes16-31 : same rows, K offset +8
    Frag16 af[2];
#pragma unroll
    for (int m = 0; m < 2; ++m) {
      const int arow = wave * 32 + m * 16 + l16;
#pragma unroll
      for (int p = 0; p < 8; ++p) {
        const int kk = ((p < 4) ? 2 * p : 2 * p + 8) + hlf * 8;
        af[m].d[p] =
            *reinterpret_cast<const unsigned int*>(&Asub[buf][arow][kk]);
      }
    }

#pragma unroll
    for (int j = 0; j < 4; ++j) {
      // B fragment (32x16): lane holds col N=l16; p0-3 -> K 0..7, p4-7 -> K 16..23; lanes16-31 K offset +8
      Frag16 bf;
#pragma unroll
      for (int p = 0; p < 8; ++p) {
        const int kk = ((p < 4) ? 2 * p : 2 * p + 8) + hlf * 8;
        bf.d[p] = *reinterpret_cast<const unsigned int*>(
            &Bsub[buf][j * 16 + l16][kk]);
      }
      acc[j * 2 + 0] = __builtin_amdgcn_wmma_f32_16x16x32_bf16(
          false, af[0].v, false, bf.v, (short)0, acc[j * 2 + 0], false, false);
      acc[j * 2 + 1] = __builtin_amdgcn_wmma_f32_16x16x32_bf16(
          false, af[1].v, false, bf.v, (short)0, acc[j * 2 + 1], false, false);
    }

    wait_async0();                       // next-tile async writes landed
    __syncthreads();                     // all waves done reading buf
  }

  // C/D layout: VGPR r -> row r (lanes 0-15) / row r+8 (lanes 16-31), col = l16
#pragma unroll
  for (int j = 0; j < 4; ++j)
#pragma unroll
    for (int m = 0; m < 2; ++m)
#pragma unroll
      for (int r = 0; r < 8; ++r) {
        const int row = mbase + wave * 32 + m * 16 + hlf * 8 + r;
        const int col = nbase + j * 16 + l16;
        const float val = acc[j * 2 + m][r];
        if (OUT_BF16)
          reinterpret_cast<unsigned short*>(Cv)[(size_t)row * N + col] =
              f2bf(val);
        else
          reinterpret_cast<float*>(Cv)[(size_t)row * N + col] = val;
      }
}

// ---------------------------------------------------------------------------
// Flash attention: grid = (SEQ/64, BATCH*NH), block = 128 threads (4 waves).
// Each wave owns 16 query rows; 64-row Q tile per block; 32-key blocks.
// qkv layout: [token][h*768 + {0..255 Q, 256..511 K, 512..767 V}], token=s*B+b.
// ---------------------------------------------------------------------------
__global__ __launch_bounds__(128)
void attn_fwd(const unsigned short* __restrict__ qkv,
              unsigned short* __restrict__ y) {
  // overlaid LDS: phase 1 uses Qs (64x260); phase 2 reuses it for Ks/Vt/Ps
  __shared__ __align__(16) unsigned short smem[19584];
  unsigned short (*Qs)[260] = reinterpret_cast<unsigned short(*)[260]>(smem);
  unsigned short (*Ks)[260] = reinterpret_cast<unsigned short(*)[260]>(smem);
  unsigned short (*Vt)[34]  =
      reinterpret_cast<unsigned short(*)[34]>(smem + 32 * 260);          // [256][34]
  unsigned short (*Ps)[40]  =
      reinterpret_cast<unsigned short(*)[40]>(smem + 32 * 260 + 256 * 34); // [64][40]

  const int qblk = blockIdx.x;          // 64 query rows per block
  const int bh   = blockIdx.y;
  const int b    = bh >> 4;
  const int h    = bh & 15;
  const int tid  = threadIdx.x;
  const int lane = tid & 31;
  const int wave = tid >> 5;
  const int hlf  = (lane >> 4) & 1;
  const int l16  = lane & 15;
  const size_t headoff = (size_t)h * 768;

  // ---- load Q tile (64 rows x 256 dims) into LDS via async copies ----------
  for (int i = tid; i < 64 * 64; i += 128) {
    const int row = i >> 6;
    const int c4  = (i & 63) * 4;       // 4-ushort (8B) chunks
    const int s   = qblk * 64 + row;
    async_ld_b64((unsigned)(uintptr_t)&Qs[row][c4],
                 qkv + (size_t)(s * BATCH + b) * F6 + headoff + c4);
  }
  wait_async0();
  __syncthreads();

  Frag16 qf[8];                          // 8 K-chunks of 32 over head dim 256
  {
    const int arow = wave * 16 + l16;
#pragma unroll
    for (int e = 0; e < 8; ++e)
#pragma unroll
      for (int p = 0; p < 8; ++p) {
        const int kk = e * 32 + ((p < 4) ? 2 * p : 2 * p + 8) + hlf * 8;
        qf[e].d[p] = *reinterpret_cast<const unsigned int*>(&Qs[arow][kk]);
      }
  }

  const float scale = 11.313708498984761f;   // sqrt(KV_CHANNELS)
  float mrow[8], lrow[8];
#pragma unroll
  for (int r = 0; r < 8; ++r) { mrow[r] = -1.0e30f; lrow[r] = 0.0f; }
  v8f o[16] = {};

  const int nkb = qblk * 2 + 2;          // 32-key blocks needed (causal)
  for (int jb = 0; jb < nkb; ++jb) {
    const int kb = jb * 32;
    __syncthreads();                     // previous iter reads done (also covers Qs)
    // ---- stage K (async, row-major) and V (manual transpose) tiles ----
    for (int i = tid; i < 32 * 64; i += 128) {
      const int row = i >> 6;
      const int c4  = (i & 63) * 4;
      const size_t base = (size_t)((kb + row) * BATCH + b) * F6 + headoff;
      async_ld_b64((unsigned)(uintptr_t)&Ks[row][c4], qkv + base + 256 + c4);
      const uint2 vv = *reinterpret_cast<const uint2*>(qkv + base + 512 + c4);
      const unsigned short* vp = reinterpret_cast<const unsigned short*>(&vv);
#pragma unroll
      for (int q = 0; q < 4; ++q) Vt[c4 + q][row] = vp[q];
    }
    wait_async0();
    __syncthreads();

    // ---- scores: S = Q * K^T  (two 16-col tiles), contraction = 256 ----
    v8f s0 = {}, s1 = {};
#pragma unroll
    for (int e = 0; e < 8; ++e) {
      Frag16 b0, b1;
#pragma unroll
      for (int p = 0; p < 8; ++p) {
        const int ee = e * 32 + ((p < 4) ? 2 * p : 2 * p + 8) + hlf * 8;
        b0.d[p] = *reinterpret_cast<const unsigned int*>(&Ks[l16][ee]);
        b1.d[p] = *reinterpret_cast<const unsigned int*>(&Ks[16 + l16][ee]);
      }
      s0 = __builtin_amdgcn_wmma_f32_16x16x32_bf16(false, qf[e].v, false, b0.v,
                                                   (short)0, s0, false, false);
      s1 = __builtin_amdgcn_wmma_f32_16x16x32_bf16(false, qf[e].v, false, b1.v,
                                                   (short)0, s1, false, false);
    }

    // ---- causal mask + online softmax (row = VGPR idx + hlf*8, col = l16) --
#pragma unroll
    for (int r = 0; r < 8; ++r) {
      const int qrow = qblk * 64 + wave * 16 + hlf * 8 + r;
      float x0 = s0[r] * scale;
      float x1 = s1[r] * scale;
      if (kb + l16 > qrow)      x0 = -3.0e38f;
      if (kb + 16 + l16 > qrow) x1 = -3.0e38f;
      float mx = fmaxf(x0, x1);
#pragma unroll
      for (int off = 8; off >= 1; off >>= 1)
        mx = fmaxf(mx, __shfl_xor(mx, off, 32));
      const float mnew = fmaxf(mrow[r], mx);
      const float p0 = __expf(x0 - mnew);
      const float p1 = __expf(x1 - mnew);
      const float corr = __expf(mrow[r] - mnew);
      float rs = p0 + p1;
#pragma unroll
      for (int off = 8; off >= 1; off >>= 1)
        rs += __shfl_xor(rs, off, 32);
      lrow[r] = lrow[r] * corr + rs;
      mrow[r] = mnew;
#pragma unroll
      for (int dd = 0; dd < 16; ++dd) o[dd][r] *= corr;
      Ps[wave * 16 + hlf * 8 + r][l16]      = f2bf(p0);
      Ps[wave * 16 + hlf * 8 + r][16 + l16] = f2bf(p1);
    }

    // ---- P (A-fragment via LDS relayout) x V (B from transposed tile) ------
    Frag16 pf;
#pragma unroll
    for (int p = 0; p < 8; ++p) {
      const int kk = ((p < 4) ? 2 * p : 2 * p + 8) + hlf * 8;
      pf.d[p] = *reinterpret_cast<const unsigned int*>(&Ps[wave * 16 + l16][kk]);
    }
#pragma unroll
    for (int dd = 0; dd < 16; ++dd) {
      Frag16 vf;
#pragma unroll
      for (int p = 0; p < 8; ++p) {
        const int kk = ((p < 4) ? 2 * p : 2 * p + 8) + hlf * 8;   // contraction = key index
        vf.d[p] = *reinterpret_cast<const unsigned int*>(&Vt[dd * 16 + l16][kk]);
      }
      o[dd] = __builtin_amdgcn_wmma_f32_16x16x32_bf16(
          false, pf.v, false, vf.v, (short)0, o[dd], false, false);
    }
  }

  // ---- normalize + write y[token][h*256 + dim] (bf16) ----
#pragma unroll
  for (int dd = 0; dd < 16; ++dd)
#pragma unroll
    for (int r = 0; r < 8; ++r) {
      const int srow = qblk * 64 + wave * 16 + hlf * 8 + r;
      const int t    = srow * BATCH + b;
      const float val = o[dd][r] / lrow[r];
      y[(size_t)t * E2 + h * 256 + dd * 16 + l16] = f2bf(val);
    }
}

// ---------------------------------------------------------------------------
extern "C" void kernel_launch(void* const* d_in, const int* in_sizes, int n_in,
                              void* d_out, int out_size, void* d_ws, size_t ws_size,
                              hipStream_t stream) {
  const float* hs    = (const float*)d_in[0];   // [SEQ][BATCH][E2]
  // d_in[1] = attention_mask (unused by reference)
  const float* wqkv  = (const float*)d_in[2];   // [F6][E2]
  const float* wproj = (const float*)d_in[3];   // [HID][E2]
  float* out = (float*)d_out;                   // [SEQ][BATCH][HID]

  unsigned short* ws = (unsigned short*)d_ws;
  size_t off = 0;
  unsigned short* hs_b    = ws + off; off += (size_t)TOK * E2;   //  32 MB
  unsigned short* wqkv_b  = ws + off; off += (size_t)F6 * E2;    //  96 MB
  unsigned short* wproj_b = ws + off; off += (size_t)HID * E2;   //  16 MB
  unsigned short* qkv_b   = ws + off; off += (size_t)TOK * F6;   //  96 MB
  unsigned short* y_b     = ws + off; off += (size_t)TOK * E2;   //  32 MB
  (void)ws_size; (void)in_sizes; (void)n_in; (void)out_size;

  cvt_f32_bf16<<<2048, 256, 0, stream>>>(hs,    hs_b,    TOK * E2);
  cvt_f32_bf16<<<4096, 256, 0, stream>>>(wqkv,  wqkv_b,  F6 * E2);
  cvt_f32_bf16<<<2048, 256, 0, stream>>>(wproj, wproj_b, HID * E2);

  // QKV projection: [TOK,E2] x [F6,E2]^T -> bf16 [TOK,F6]
  gemm_bf16<true><<<dim3(F6 / 64, TOK / 128), 128, 0, stream>>>(
      hs_b, wqkv_b, qkv_b, TOK, F6, E2);

  // Flash attention -> bf16 y [TOK,E2]
  attn_fwd<<<dim3(SEQ / 64, BATCH * NH), 128, 0, stream>>>(qkv_b, y_b);

  // Output projection: [TOK,E2] x [HID,E2]^T -> fp32 d_out
  gemm_bf16<false><<<dim3(HID / 64, TOK / 128), 128, 0, stream>>>(
      y_b, wproj_b, out, TOK, HID, E2);
}
